// ABS_LSTM4_62586263437986
// MI455X (gfx1250) — matrix operand order repaired and checked by reading, not compiled, not run on hardware
//
#include <hip/hip_runtime.h>

#define BB 128
#define TT 512
#define DD 300
#define UU 256
#define GG 1024
#define KPX 320              // D padded to multiple of 32
#define ROWS (BB*TT)         // 65536

typedef __attribute__((ext_vector_type(16))) __bf16 v16bf;
typedef __attribute__((ext_vector_type(8)))  __bf16 v8bf;
typedef __attribute__((ext_vector_type(8)))  float  v8f;

// ---------- per-direction state block offsets (bytes, all 256-aligned) ----------
constexpr size_t OFF_HR   = 0;                     // hRing   f32 [3][128][256]
constexpr size_t OFF_HRB  = 393216;                // hRingBf bf16
constexpr size_t OFF_CR   = 589824;                // cRing   f32
constexpr size_t OFF_CRB  = 983040;                // cRingBf bf16
constexpr size_t OFF_HWR  = 1179648;               // hWRing  f32 [3][128][256]
constexpr size_t OFF_CWR  = 1572864;               // cWRing  f32 [3][128][16]
constexpr size_t OFF_HABF = 1597440;               // haBf    bf16 [128][256]
constexpr size_t OFF_CAF  = 1662976;               // caF     f32  [128][256]
constexpr size_t OFF_GBUF = 1794048;               // gBuf    f32  [128][1024]
constexpr size_t OFF_BARS = 2318336;               // bars    int  [512*4]
constexpr size_t DIR_STRIDE = 2326528;

// ---------------------------- small device helpers ----------------------------
__device__ __forceinline__ v8f v8f_zero() {
  v8f z;
#pragma unroll
  for (int i = 0; i < 8; ++i) z[i] = 0.0f;
  return z;
}

// 16x32 bf16 A-fragment from row-major bf16 (ld elements per row).
// lanes 0-15: row=lane, K = 0..7 and 16..23 ; lanes 16-31: row=lane-16, K = 8..15 and 24..31
__device__ __forceinline__ v16bf load_a_frag(const __bf16* p) {
  v8bf lo = *(const v8bf*)p;
  v8bf hi = *(const v8bf*)(p + 16);
  v16bf r;
#pragma unroll
  for (int i = 0; i < 8; ++i) { r[i] = lo[i]; r[i + 8] = hi[i]; }
  return r;
}

__device__ __forceinline__ v8f wmma_bf16(v16bf a, v16bf b, v8f c) {
  return __builtin_amdgcn_wmma_f32_16x16x32_bf16(false, a, false, b, (short)0, c,
                                                 false, false);
}

__device__ __forceinline__ float hsigf(float x) {
  return fminf(fmaxf(0.2f * x + 0.5f, 0.0f), 1.0f);
}
__device__ __forceinline__ float eluf(float x) {
  return x > 0.0f ? x : (expf(x) - 1.0f);
}

__device__ __forceinline__ void grid_barrier(int* bar, int nBlocks) {
  __threadfence();
  __syncthreads();
  if (threadIdx.x == 0) {
    atomicAdd(bar, 1);
    while (__hip_atomic_load(bar, __ATOMIC_ACQUIRE, __HIP_MEMORY_SCOPE_AGENT) < nBlocks) {
      __builtin_amdgcn_s_sleep(2);
    }
  }
  __syncthreads();
}

// ---------------------------------- kernels -----------------------------------
__global__ void zero_kernel(unsigned int* p, size_t n) {
  for (size_t i = blockIdx.x * (size_t)blockDim.x + threadIdx.x; i < n;
       i += (size_t)gridDim.x * blockDim.x)
    p[i] = 0u;
}

// f32 -> bf16 with K zero-padding (row-major)
__global__ void pack_x(const float* __restrict__ src, __bf16* __restrict__ dst,
                       int rows, int K, int Kp) {
  size_t n = (size_t)rows * Kp;
  for (size_t i = blockIdx.x * (size_t)blockDim.x + threadIdx.x; i < n;
       i += (size_t)gridDim.x * blockDim.x) {
    size_t r = i / Kp;
    int k = (int)(i % Kp);
    dst[i] = (k < K) ? (__bf16)src[r * K + k] : (__bf16)0.0f;
  }
}

// Pack row-major f32 KxN weight into WMMA B-fragment order, zero-padded.
// Tile (kt,nt): lane<16 -> col=nt*16+lane, K=kt*32+0..7,16..23 ; lane>=16 -> K=kt*32+8..15,24..31
__global__ void pack_b(const float* __restrict__ src, __bf16* __restrict__ dst,
                       int K, int N, int ld, int KT, int NT) {
  int idx = blockIdx.x * blockDim.x + threadIdx.x;
  int total = KT * NT * 32;
  if (idx >= total) return;
  int lane = idx & 31;
  int tile = idx >> 5;
  int nt = tile % NT, kt = tile / NT;
  int n = nt * 16 + (lane & 15);
  int k0 = kt * 32 + ((lane >> 4) ? 8 : 0);
  __bf16* d = dst + (size_t)tile * 512 + lane * 16;
#pragma unroll
  for (int j = 0; j < 16; ++j) {
    int k = k0 + (j < 8 ? j : j + 8);
    float v = (k < K && n < N) ? src[(size_t)k * ld + n] : 0.0f;
    d[j] = (__bf16)v;
  }
}

// C = A_bf16 (row-major, lda) x Bp (packed fragments). NW tiles per wave.
// PERM=true: input row m = b*TT+t is stored at output row t*BB+b (time-major out).
template <typename OutT, int NW, bool PERM>
__global__ __launch_bounds__(256) void gemm_wmma(const __bf16* __restrict__ A,
                                                 const __bf16* __restrict__ Bp,
                                                 OutT* __restrict__ C, int MT, int NT,
                                                 int KT, int lda, int ldc) {
  int wave = (blockIdx.x * blockDim.x + threadIdx.x) >> 5;
  int lane = threadIdx.x & 31;
  int groups = NT / NW;
  int mt = wave / groups;
  int ng = wave % groups;
  if (mt >= MT) return;
  v8f acc[NW];
#pragma unroll
  for (int j = 0; j < NW; ++j) acc[j] = v8f_zero();
  int rowA = lane & 15, koff = (lane >> 4) * 8;
  const __bf16* aP = A + ((size_t)(mt * 16 + rowA)) * lda + koff;
  for (int kt = 0; kt < KT; ++kt) {
    v16bf a = load_a_frag(aP);
    aP += 32;
#pragma unroll
    for (int j = 0; j < NW; ++j) {
      v16bf b = ((const v16bf*)Bp)[((size_t)kt * NT + ng * NW + j) * 32 + lane];
      acc[j] = wmma_bf16(a, b, acc[j]);
    }
  }
  int rowBase = mt * 16 + ((lane >> 4) * 8);
  int col = lane & 15;
#pragma unroll
  for (int j = 0; j < NW; ++j) {
    int n0 = (ng * NW + j) * 16 + col;
#pragma unroll
    for (int r = 0; r < 8; ++r) {
      size_t m = (size_t)(rowBase + r);
      size_t mrow = PERM ? ((m % TT) * BB + (m / TT)) : m;
      C[mrow * ldc + n0] = (OutT)acc[j][r];
    }
  }
}

// Persistent bidirectional recurrent kernel. 2 dirs x 32 blocks x 256 threads.
// Z/XH/XC are time-major: [(t*BB + b) * width].
__global__ __launch_bounds__(256, 1) void recurrent_kernel(
    const __bf16* __restrict__ Z, const __bf16* __restrict__ XH,
    const __bf16* __restrict__ XC, const __bf16* __restrict__ Rp,
    const __bf16* __restrict__ WhP, const __bf16* __restrict__ WcP,
    float* __restrict__ hsF, __bf16* __restrict__ hsFbf, float* __restrict__ hsB,
    __bf16* __restrict__ hsBbf, char* __restrict__ stateBase, size_t dirStride) {
  const int NB = 32;
  int dir = blockIdx.x / NB;
  int lb = blockIdx.x % NB;
  char* st = stateBase + (size_t)dir * dirStride;
  float*  hRing   = (float*)(st + OFF_HR);
  __bf16* hRingBf = (__bf16*)(st + OFF_HRB);
  float*  cRing   = (float*)(st + OFF_CR);
  __bf16* cRingBf = (__bf16*)(st + OFF_CRB);
  float*  hWRing  = (float*)(st + OFF_HWR);
  float*  cWRing  = (float*)(st + OFF_CWR);
  __bf16* haBf    = (__bf16*)(st + OFF_HABF);
  float*  caF     = (float*)(st + OFF_CAF);
  float*  gBuf    = (float*)(st + OFF_GBUF);
  int*    bars    = (int*)(st + OFF_BARS);
  float*  hs      = dir ? hsB : hsF;
  __bf16* hsbf    = dir ? hsBbf : hsFbf;

  int tid = threadIdx.x, lane = tid & 31, wid = tid >> 5;
  int bOwn = lb * 4;  // each block owns 4 batch rows

  __shared__ float sS[4][2][3];
  __shared__ float sW[4][2][3];

  for (int s = 0; s < TT; ++s) {
    int t = dir ? (TT - 1 - s) : s;
    int tn = dir ? (t - 1) : (t + 1);   // next step's time index
    int slots[3];
#pragma unroll
    for (int j = 0; j < 3; ++j) slots[j] = ((s - 1 - j) % 3 + 3) % 3;
    int slotW = s % 3;

    // ---- Phase 1: scores + softmax + ha/ca (block-local rows) ----
    for (int task = wid; task < 24; task += 8) {
      int isC = task >= 12 ? 1 : 0;
      int tt2 = isC ? task - 12 : task;
      int r = tt2 / 3, j = tt2 % 3;
      int b = bOwn + r;
      int slot = slots[j];
      float sum = 0.0f;
      if (!isC) {
        const float* hw = hWRing + ((size_t)slot * BB + b) * UU;
        const __bf16* xh = XH + ((size_t)t * BB + b) * UU;
        for (int u = lane; u < UU; u += 32) {
          float v = tanhf(hw[u] + (float)xh[u]);
          sum += v * v;
        }
      } else if (lane < 5) {
        const float* cw = cWRing + ((size_t)slot * BB + b) * 16;
        const __bf16* xc = XC + ((size_t)t * BB + b) * 16;
        float v = tanhf(cw[lane] + (float)xc[lane]);
        sum = v * v;
      }
#pragma unroll
      for (int m = 16; m >= 1; m >>= 1) sum += __shfl_xor(sum, m, 32);
      if (lane == 0) sS[r][isC][j] = sqrtf(sum);
    }
    __syncthreads();
    if (tid < 8) {
      int r = tid >> 1, w = tid & 1;
      float a0 = sS[r][w][0], a1 = sS[r][w][1], a2 = sS[r][w][2];
      float mx = fmaxf(a0, fmaxf(a1, a2));
      float e0 = expf(a0 - mx), e1 = expf(a1 - mx), e2 = expf(a2 - mx);
      float inv = 1.0f / (e0 + e1 + e2);
      sW[r][w][0] = e0 * inv; sW[r][w][1] = e1 * inv; sW[r][w][2] = e2 * inv;
    }
    // prefetch next step's XH block for owned rows (4 rows x 512B)
    if (tn >= 0 && tn < TT && tid < 16) {
      int r = tid >> 2, chunk = tid & 3;
      __builtin_prefetch(XH + ((size_t)tn * BB + bOwn + r) * UU + chunk * 64, 0, 1);
    }
    __syncthreads();
    for (int i = tid; i < 4 * UU; i += 256) {
      int r = i >> 8, u = i & 255;
      int b = bOwn + r;
      size_t i0 = ((size_t)slots[0] * BB + b) * UU + u;
      size_t i1 = ((size_t)slots[1] * BB + b) * UU + u;
      size_t i2 = ((size_t)slots[2] * BB + b) * UU + u;
      float ha = fmaxf(sW[r][0][0] * hRing[i0] + sW[r][0][1] * hRing[i1] +
                           sW[r][0][2] * hRing[i2], 0.0f);
      haBf[(size_t)b * UU + u] = (__bf16)ha;
      float ca = fmaxf(sW[r][1][0] * cRing[i0] + sW[r][1][1] * cRing[i1] +
                           sW[r][1][2] * cRing[i2], 0.0f);
      caF[(size_t)b * UU + u] = ca;
    }
    grid_barrier(bars + s * 4 + 0, NB);

    // ---- Phase 2: g = hsig(Z[t] + ha @ R)  (M=128,N=1024,K=256) ----
    {
      int w = lb * 8 + wid;  // 0..255 waves
      int mt = w >> 5;       // 0..7
      int ng = w & 31;       // 2 N-tiles each
      v8f acc0 = v8f_zero(), acc1 = v8f_zero();
      int rowA = lane & 15, koff = (lane >> 4) * 8;
      const __bf16* aP = haBf + ((size_t)(mt * 16 + rowA)) * UU + koff;
      for (int kt = 0; kt < 8; ++kt) {
        v16bf a = load_a_frag(aP);
        aP += 32;
        v16bf b0 = ((const v16bf*)Rp)[((size_t)kt * 64 + ng * 2 + 0) * 32 + lane];
        v16bf b1v = ((const v16bf*)Rp)[((size_t)kt * 64 + ng * 2 + 1) * 32 + lane];
        acc0 = wmma_bf16(a, b0, acc0);
        acc1 = wmma_bf16(a, b1v, acc1);
      }
      // prefetch next step's Z region covered by this wave (16 rows x 64B)
      if (tn >= 0 && tn < TT) {
        const __bf16* zn = Z + ((size_t)tn * BB + mt * 16) * GG + ng * 32;
        int rr = lane >> 1, half = lane & 1;
        __builtin_prefetch(zn + (size_t)rr * GG + half * 16, 0, 1);
      }
      int rowBase = mt * 16 + ((lane >> 4) * 8);
      int col = lane & 15;
#pragma unroll
      for (int j = 0; j < 2; ++j) {
        v8f acc = j ? acc1 : acc0;
        int n = (ng * 2 + j) * 16 + col;
#pragma unroll
        for (int r = 0; r < 8; ++r) {
          int b = rowBase + r;
          float z = (float)Z[((size_t)t * BB + b) * GG + n];
          gBuf[(size_t)b * GG + n] = hsigf(acc[r] + z);
        }
      }
    }
    grid_barrier(bars + s * 4 + 1, NB);

    // ---- Phase 3: LSTM state update for owned rows ----
    for (int i = tid; i < 4 * UU; i += 256) {
      int r = i >> 8, u = i & 255;
      int b = bOwn + r;
      const float* g = gBuf + (size_t)b * GG;
      float gi = g[u], gf = g[UU + u], gm = g[2 * UU + u], go = g[3 * UU + u];
      float c = gf * caF[(size_t)b * UU + u] + gi * gm;
      float h = go * tanhf(c);
      size_t ri = ((size_t)slotW * BB + b) * UU + u;
      hRing[ri] = h;  hRingBf[ri] = (__bf16)h;
      cRing[ri] = c;  cRingBf[ri] = (__bf16)c;
      size_t oi = ((size_t)b * TT + s) * UU + u;
      hs[oi] = h;     hsbf[oi] = (__bf16)h;
    }
    grid_barrier(bars + s * 4 + 2, NB);

    // ---- Phase 4: hW_new = h_new @ WhHead (8x16 tiles), cW_new = c_new @ WcHead ----
    {
      int w = lb * 8 + wid;
      int rowA = lane & 15, koff = (lane >> 4) * 8;
      if (w < 128) {
        int mt = w >> 4, nt = w & 15;
        v8f acc = v8f_zero();
        const __bf16* aP =
            hRingBf + (size_t)slotW * BB * UU + ((size_t)(mt * 16 + rowA)) * UU + koff;
        for (int kt = 0; kt < 8; ++kt) {
          v16bf a = load_a_frag(aP);
          aP += 32;
          v16bf b = ((const v16bf*)WhP)[((size_t)kt * 16 + nt) * 32 + lane];
          acc = wmma_bf16(a, b, acc);
        }
        int rowBase = mt * 16 + ((lane >> 4) * 8);
        int col = lane & 15;
#pragma unroll
        for (int r = 0; r < 8; ++r)
          hWRing[((size_t)slotW * BB + rowBase + r) * UU + nt * 16 + col] = acc[r];
      } else if (w < 136) {
        int mt = w - 128;
        v8f acc = v8f_zero();
        const __bf16* aP =
            cRingBf + (size_t)slotW * BB * UU + ((size_t)(mt * 16 + rowA)) * UU + koff;
        for (int kt = 0; kt < 8; ++kt) {
          v16bf a = load_a_frag(aP);
          aP += 32;
          v16bf b = ((const v16bf*)WcP)[(size_t)kt * 32 + lane];
          acc = wmma_bf16(a, b, acc);
        }
        int rowBase = mt * 16 + ((lane >> 4) * 8);
        int col = lane & 15;
#pragma unroll
        for (int r = 0; r < 8; ++r)
          cWRing[((size_t)slotW * BB + rowBase + r) * 16 + col] = acc[r];
      }
    }
    grid_barrier(bars + s * 4 + 3, NB);
  }
}

// Intra attention: a[row] = sum_u hsig(hs@A1out + xA1 + b1)[row,u] * A2[u]
__global__ __launch_bounds__(256) void intra_gemm(const __bf16* __restrict__ hsbf,
                                                  const __bf16* __restrict__ Bp,
                                                  const __bf16* __restrict__ xA1,
                                                  const float* __restrict__ b1,
                                                  const float* __restrict__ a2,
                                                  float* __restrict__ aOut) {
  int wave = (blockIdx.x * blockDim.x + threadIdx.x) >> 5;
  int lane = threadIdx.x & 31;
  int mt = wave >> 2;
  int ng = wave & 3;
  if (mt >= ROWS / 16) return;
  v8f acc[4];
#pragma unroll
  for (int j = 0; j < 4; ++j) acc[j] = v8f_zero();
  int rowA = lane & 15, koff = (lane >> 4) * 8;
  const __bf16* aP = hsbf + ((size_t)(mt * 16 + rowA)) * UU + koff;
  for (int kt = 0; kt < 8; ++kt) {
    v16bf a = load_a_frag(aP);
    aP += 32;
#pragma unroll
    for (int j = 0; j < 4; ++j) {
      v16bf b = ((const v16bf*)Bp)[((size_t)kt * 16 + ng * 4 + j) * 32 + lane];
      acc[j] = wmma_bf16(a, b, acc[j]);
    }
  }
  int rowBase = mt * 16 + ((lane >> 4) * 8);
  int col = lane & 15;
  float part[8];
#pragma unroll
  for (int r = 0; r < 8; ++r) part[r] = 0.0f;
#pragma unroll
  for (int j = 0; j < 4; ++j) {
    int u = (ng * 4 + j) * 16 + col;
    float w2 = a2[u];
    float bb = b1[u];
#pragma unroll
    for (int r = 0; r < 8; ++r) {
      size_t row = (size_t)rowBase + r;
      float v = acc[j][r] + (float)xA1[row * UU + u] + bb;
      part[r] += hsigf(v) * w2;
    }
  }
#pragma unroll
  for (int r = 0; r < 8; ++r) {
    float p = part[r];
#pragma unroll
    for (int m = 8; m >= 1; m >>= 1) p += __shfl_xor(p, m, 16);
    if ((lane & 15) == 0) atomicAdd(&aOut[rowBase + r], p);
  }
}

__global__ void final_out(const float* __restrict__ hsF, const float* __restrict__ aF,
                          const float* __restrict__ hsB, const float* __restrict__ aB,
                          float* __restrict__ out) {
  size_t n = (size_t)ROWS * UU;
  for (size_t i = blockIdx.x * (size_t)blockDim.x + threadIdx.x; i < n;
       i += (size_t)gridDim.x * blockDim.x) {
    size_t r = i >> 8;
    int u = (int)(i & 255);
    size_t b = r >> 9;
    int tt = (int)(r & 511);
    float vf = eluf(aF[r] * hsF[i]);
    size_t rb = (b << 9) + (TT - 1 - tt);
    float vb = eluf(aB[rb] * hsB[(rb << 8) + u]);
    size_t ob = ((b * TT + tt) * (size_t)(2 * UU));
    out[ob + u] = vf;
    out[ob + UU + u] = vb;
  }
}

// ---------------------------------- launcher -----------------------------------
extern "C" void kernel_launch(void* const* d_in, const int* in_sizes, int n_in,
                              void* d_out, int out_size, void* d_ws, size_t ws_size,
                              hipStream_t stream) {
  (void)in_sizes; (void)n_in; (void)out_size; (void)ws_size;
  const float* x    = (const float*)d_in[0];
  const float* ker  = (const float*)d_in[1];
  const float* rker = (const float*)d_in[2];
  const float* ah   = (const float*)d_in[3];
  const float* ac   = (const float*)d_in[4];
  const float* a1   = (const float*)d_in[5];
  const float* b1   = (const float*)d_in[6];
  const float* a2   = (const float*)d_in[7];
  float* out = (float*)d_out;

  char* base = (char*)d_ws;
  size_t off = 0;
  auto alloc = [&](size_t bytes) -> char* {
    char* p = base + off;
    off += (bytes + 255) & ~(size_t)255;
    return p;
  };

  __bf16* xbf  = (__bf16*)alloc((size_t)ROWS * KPX * 2);
  __bf16* kerP = (__bf16*)alloc((size_t)10 * 64 * 512 * 2);
  __bf16* rP   = (__bf16*)alloc((size_t)8 * 64 * 512 * 2);
  __bf16* whP  = (__bf16*)alloc((size_t)8 * 16 * 512 * 2);
  __bf16* ahxP = (__bf16*)alloc((size_t)10 * 16 * 512 * 2);
  __bf16* wcP  = (__bf16*)alloc((size_t)8 * 1 * 512 * 2);
  __bf16* wcxP = (__bf16*)alloc((size_t)10 * 1 * 512 * 2);
  __bf16* a1oP = (__bf16*)alloc((size_t)8 * 16 * 512 * 2);
  __bf16* a1xP = (__bf16*)alloc((size_t)10 * 16 * 512 * 2);
  __bf16* Zb   = (__bf16*)alloc((size_t)ROWS * GG * 2);   // time-major
  __bf16* XHb  = (__bf16*)alloc((size_t)ROWS * UU * 2);   // time-major
  __bf16* XCb  = (__bf16*)alloc((size_t)ROWS * 16 * 2);   // time-major
  __bf16* xA1b = (__bf16*)alloc((size_t)ROWS * UU * 2);   // batch-major
  float*  hsF  = (float*)alloc((size_t)ROWS * UU * 4);
  __bf16* hsFb = (__bf16*)alloc((size_t)ROWS * UU * 2);
  float*  hsB  = (float*)alloc((size_t)ROWS * UU * 4);
  __bf16* hsBb = (__bf16*)alloc((size_t)ROWS * UU * 2);
  size_t zeroStart = off;
  float* aF = (float*)alloc((size_t)ROWS * 4);
  float* aB = (float*)alloc((size_t)ROWS * 4);
  char* state = alloc(2 * DIR_STRIDE);
  size_t zeroBytes = off - zeroStart;

  // zero barriers / rings / accumulators
  zero_kernel<<<2048, 256, 0, stream>>>((unsigned int*)(base + zeroStart),
                                        zeroBytes / 4);
  // pack activations + weights
  pack_x<<<4096, 256, 0, stream>>>(x, xbf, ROWS, DD, KPX);
  auto packW = [&](const float* src, __bf16* dst, int K, int N, int ld, int KT,
                   int NT) {
    int tot = KT * NT * 32;
    pack_b<<<(tot + 255) / 256, 256, 0, stream>>>(src, dst, K, N, ld, KT, NT);
  };
  packW(ker, kerP, DD, GG, GG, 10, 64);
  packW(rker, rP, UU, GG, GG, 8, 64);
  packW(ah, whP, UU, UU, UU, 8, 16);                       // attention_h head rows
  packW(ah + (size_t)UU * UU, ahxP, DD, UU, UU, 10, 16);   // attention_h x rows
  packW(ac, wcP, UU, 5, 5, 8, 1);                          // attention_c head rows
  packW(ac + (size_t)UU * 5, wcxP, DD, 5, 5, 10, 1);       // attention_c x rows
  packW(a1, a1oP, UU, UU, UU, 8, 16);                      // attention1 out rows
  packW(a1 + (size_t)UU * UU, a1xP, DD, UU, UU, 10, 16);   // attention1 x rows

  // precompute GEMMs (bf16 out; Z/XH/XC permuted to time-major)
  gemm_wmma<__bf16, 4, true><<<8192, 256, 0, stream>>>(xbf, kerP, Zb, ROWS / 16, 64,
                                                       10, KPX, GG);
  gemm_wmma<__bf16, 4, true><<<2048, 256, 0, stream>>>(xbf, ahxP, XHb, ROWS / 16, 16,
                                                       10, KPX, UU);
  gemm_wmma<__bf16, 1, true><<<512, 256, 0, stream>>>(xbf, wcxP, XCb, ROWS / 16, 1,
                                                      10, KPX, 16);
  gemm_wmma<__bf16, 4, false><<<2048, 256, 0, stream>>>(xbf, a1xP, xA1b, ROWS / 16,
                                                        16, 10, KPX, UU);

  // sequential bidirectional recurrence (persistent, grid-sync per phase)
  recurrent_kernel<<<64, 256, 0, stream>>>(Zb, XHb, XCb, rP, whP, wcP, hsF, hsFb,
                                           hsB, hsBb, state, (size_t)DIR_STRIDE);

  // intra attention (fused GEMM + hsig + dot(attention2))
  intra_gemm<<<2048, 256, 0, stream>>>(hsFb, a1oP, xA1b, b1, a2, aF);
  intra_gemm<<<2048, 256, 0, stream>>>(hsBb, a1oP, xA1b, b1, a2, aB);

  // elu(a * outs), backward time-reversal, concat into (B,T,2U)
  final_out<<<4096, 256, 0, stream>>>(hsF, aF, hsB, aB, out);
}
